// DecomposingAttnProcessor_41669772706190
// MI455X (gfx1250) — matrix operand order repaired and checked by reading, not compiled, not run on hardware
//
#include <hip/hip_runtime.h>
#include <math.h>

// ---------------------------------------------------------------------------
// Types for CDNA5 WMMA (wave32): v_wmma_f32_16x16x32_bf16
// ---------------------------------------------------------------------------
typedef __attribute__((ext_vector_type(16))) __bf16 v16bf;
typedef __attribute__((ext_vector_type(8)))  float  v8f;
typedef __attribute__((ext_vector_type(4)))  unsigned int u32x4;
typedef __attribute__((ext_vector_type(4)))  float  f32x4;
typedef __attribute__((ext_vector_type(8)))  unsigned short u16x8;

union FragU { v16bf bf; u32x4 u[2]; };

__device__ __forceinline__ unsigned short f2bf(float f) {
  unsigned int u = __builtin_bit_cast(unsigned int, f);
  u += 0x7FFFu + ((u >> 16) & 1u);                 // round-to-nearest-even
  return (unsigned short)(u >> 16);
}

__device__ __forceinline__ v8f wmma_bf16(v16bf a, v16bf b, v8f c) {
  // 8 args: (neg_a, A, neg_b, B, c_mod, C, reuse_a, reuse_b)
  return __builtin_amdgcn_wmma_f32_16x16x32_bf16(false, a, false, b, (short)0, c,
                                                 false, false);
}

// A fragment 16x32 bf16 from row-major memory (row stride ldk elements).
// ISA layout: lanes 0-15 -> M=lane, k 0..7 & 16..23; lanes 16-31 -> k 8..15 & 24..31.
__device__ __forceinline__ v16bf load_a(const unsigned short* base, int ldk, int lane) {
  int row  = lane & 15;
  int koff = (lane >> 4) * 8;
  const unsigned short* p = base + (size_t)row * ldk + koff;
  FragU f;
  f.u[0] = *(const u32x4*)(p);        // k koff..koff+7
  f.u[1] = *(const u32x4*)(p + 16);   // k koff+16..koff+23
  return f.bf;
}

// B fragment 32x16 bf16 from B^T row-major [N][K]: lane n = lane&15,
// k koff..koff+7 & koff+16..koff+23 with koff = 8*(lane>>4) (same as A).
__device__ __forceinline__ v16bf load_b(const unsigned short* base, int ldk, int lane) {
  int n    = lane & 15;
  int koff = (lane >> 4) * 8;
  const unsigned short* p = base + (size_t)n * ldk + koff;
  FragU f;
  f.u[0] = *(const u32x4*)(p);
  f.u[1] = *(const u32x4*)(p + 16);
  return f.bf;
}

// ---------------------------------------------------------------------------
// f32 -> bf16 conversion kernels
// ---------------------------------------------------------------------------
// Vectorized: 8 elements/thread (2x b128 load, 1x b128 store). n8 = n/8.
__global__ void cvt_bf16_x8(const float* __restrict__ src,
                            unsigned short* __restrict__ dst, int n8) {
  int i = blockIdx.x * blockDim.x + threadIdx.x;
  int stride = gridDim.x * blockDim.x;
  for (; i < n8; i += stride) {
    const f32x4* s = (const f32x4*)src + (size_t)i * 2;
    f32x4 a = s[0], b = s[1];
    u16x8 o;
#pragma unroll
    for (int v = 0; v < 4; ++v) { o[v] = f2bf(a[v]); o[4 + v] = f2bf(b[v]); }
    *((u16x8*)dst + i) = o;
  }
}

// Convert and transpose [K][N] f32 -> [N][K] bf16 (weights -> B^T layout).
__global__ void cvt_bf16_T(const float* __restrict__ src,
                           unsigned short* __restrict__ dst, int K, int N) {
  int i = blockIdx.x * blockDim.x + threadIdx.x;
  int stride = gridDim.x * blockDim.x;
  int total = K * N;
  for (; i < total; i += stride) {
    int k = i / N, n = i - k * N;
    dst[(size_t)n * K + k] = f2bf(src[i]);
  }
}

// v_bf [bc][se][h*64+dh] -> vT [bc][h][dh][se]   (8*256*1024 elements)
__global__ void transpose_v(const unsigned short* __restrict__ v,
                            unsigned short* __restrict__ vT) {
  int i = blockIdx.x * blockDim.x + threadIdx.x;
  if (i >= 8 * 256 * 1024) return;
  int bc = i >> 18;
  int se = (i >> 10) & 255;
  int hd = i & 1023;
  int h = hd >> 6, dh = hd & 63;
  vT[((((size_t)bc * 16 + h) * 64 + dh) << 8) + se] = v[i];
}

// ---------------------------------------------------------------------------
// bf16 GEMM core: wave tile 64x64 (acc[4][4]); 16 b128 loads per 16 WMMAs
// per k-step. Unconditional prefetch two k-iterations ahead (speculative
// global_prefetch_b8 — OOB prefetches are silently dropped, no counters).
// ---------------------------------------------------------------------------
__device__ __forceinline__ void gemm_core(
    const unsigned short* __restrict__ A, const unsigned short* __restrict__ BT,
    int K, int m0, int n0, int lane, v8f acc[4][4]) {
#pragma unroll
  for (int i = 0; i < 4; ++i)
#pragma unroll
    for (int j = 0; j < 4; ++j) acc[i][j] = (v8f)(0.0f);

  const unsigned short* pfa  = A + (size_t)(m0 + lane) * K + 64;
  const unsigned short* pfa2 = A + (size_t)(m0 + 32 + lane) * K + 64;
  const unsigned short* pfb  = BT + (size_t)(n0 + lane) * K + 64;
  const unsigned short* pfb2 = BT + (size_t)(n0 + 32 + lane) * K + 64;

  for (int k = 0; k < K; k += 32) {
    __builtin_prefetch(pfa + k, 0, 1);
    __builtin_prefetch(pfa2 + k, 0, 1);
    __builtin_prefetch(pfb + k, 0, 1);
    __builtin_prefetch(pfb2 + k, 0, 1);
    v16bf a[4], b[4];
#pragma unroll
    for (int i = 0; i < 4; ++i)
      a[i] = load_a(A + (size_t)(m0 + i * 16) * K + k, K, lane);
#pragma unroll
    for (int j = 0; j < 4; ++j)
      b[j] = load_b(BT + (size_t)(n0 + j * 16) * K + k, K, lane);
#pragma unroll
    for (int i = 0; i < 4; ++i)
#pragma unroll
      for (int j = 0; j < 4; ++j) acc[i][j] = wmma_bf16(a[i], b[j], acc[i][j]);
  }
}

// 256 threads = 8 waves (2m x 4n); block tile 128x256. bf16 output, plain
// cached stores (consumed by the next kernel; may stay resident in 192MB L2).
__global__ __launch_bounds__(256) void gemm_bf16_out(
    const unsigned short* __restrict__ A, const unsigned short* __restrict__ BT,
    unsigned short* __restrict__ C, int M, int N, int K) {
  int lane = threadIdx.x & 31;
  int wave = threadIdx.x >> 5;
  int m0 = blockIdx.y * 128 + (wave >> 2) * 64;
  int n0 = blockIdx.x * 256 + (wave & 3) * 64;

  v8f acc[4][4];
  gemm_core(A, BT, K, m0, n0, lane, acc);

  int hl = lane >> 4, nl = lane & 15;
#pragma unroll
  for (int i = 0; i < 4; ++i)
#pragma unroll
    for (int j = 0; j < 4; ++j)
#pragma unroll
      for (int v = 0; v < 8; ++v) {
        int m = m0 + i * 16 + v + 8 * hl;
        int n = n0 + j * 16 + nl;
        C[(size_t)m * N + n] = f2bf(acc[i][j][v]);
      }
}

// f32 output with fused bias + residual (final projection). Output is never
// re-read on device -> non-temporal stores; residual is read-once -> NT loads.
__global__ __launch_bounds__(256) void gemm_f32_epi(
    const unsigned short* __restrict__ A, const unsigned short* __restrict__ BT,
    float* __restrict__ C, const float* __restrict__ bias,
    const float* __restrict__ resid, int M, int N, int K) {
  int lane = threadIdx.x & 31;
  int wave = threadIdx.x >> 5;
  int m0 = blockIdx.y * 128 + (wave >> 2) * 64;
  int n0 = blockIdx.x * 256 + (wave & 3) * 64;

  v8f acc[4][4];
  gemm_core(A, BT, K, m0, n0, lane, acc);

  int hl = lane >> 4, nl = lane & 15;
  float bs[4];
#pragma unroll
  for (int j = 0; j < 4; ++j) bs[j] = bias[n0 + j * 16 + nl];
#pragma unroll
  for (int i = 0; i < 4; ++i)
#pragma unroll
    for (int j = 0; j < 4; ++j)
#pragma unroll
      for (int v = 0; v < 8; ++v) {
        int m = m0 + i * 16 + v + 8 * hl;
        int n = n0 + j * 16 + nl;
        size_t idx = (size_t)m * N + n;
        float r = __builtin_nontemporal_load(resid + idx);
        __builtin_nontemporal_store(acc[i][j][v] + bs[j] + r, C + idx);
      }
}

// ---------------------------------------------------------------------------
// Fused attention: scores -> 4-way component softmax -> entropy -> w @ V.
// grid (s_block=64, h=16, b=2); 256 threads = 8 waves; wave -> (component c,
// 32-row sub-tile). Streams Se in tiles of 32 (softmax is across components,
// elementwise in (s,t), so no row reductions over Se are needed).
// ---------------------------------------------------------------------------
__global__ __launch_bounds__(256) void fused_attn(
    const unsigned short* __restrict__ Q,   // [8][4096][1024]
    const unsigned short* __restrict__ Kb,  // [8][256][1024]
    const unsigned short* __restrict__ VT,  // [8][16][64][256]
    unsigned short* __restrict__ O,         // [8][4096][1024] bf16
    float* __restrict__ ENT)                // [2][16][4096][256]
{
  __shared__ __align__(16) float          sS[4][64][32];
  __shared__ __align__(16) unsigned short sW[4][64][32];

  int tid  = threadIdx.x;
  int lane = tid & 31;
  int wave = tid >> 5;
  int c    = wave >> 1;     // component 0..3
  int msub = wave & 1;      // 32-row sub-tile

  int sblk = blockIdx.x;
  int h    = blockIdx.y;
  int b    = blockIdx.z;
  int bc   = c * 2 + b;     // reshape(C,B,...) => bc = c*B + b
  int s0   = sblk * 64;

  const unsigned short* qbase = Q + ((size_t)bc * 4096 + s0 + msub * 32) * 1024 + h * 64;
  const unsigned short* kbase = Kb + (size_t)bc * 256 * 1024 + h * 64;
  const unsigned short* vbase = VT + ((size_t)bc * 16 + h) * 64 * 256;

  const float scale  = 0.125f;          // dh^-0.5, dh=64
  const float wscale = 4.0f / 256.0f;   // C/Se

  v8f oacc[2][4];
#pragma unroll
  for (int i = 0; i < 2; ++i)
#pragma unroll
    for (int j = 0; j < 4; ++j) oacc[i][j] = (v8f)(0.0f);

  int hl = lane >> 4, nl = lane & 15;

  for (int t0 = 0; t0 < 256; t0 += 32) {
    // ---- scores: S_c = Q_c(32x64) * K_c^T(64x32) ----
    v8f sacc[2][2];
#pragma unroll
    for (int i = 0; i < 2; ++i)
#pragma unroll
      for (int j = 0; j < 2; ++j) sacc[i][j] = (v8f)(0.0f);

#pragma unroll
    for (int ks = 0; ks < 2; ++ks) {
      int kk = ks * 32;
      v16bf a[2], bfr[2];
#pragma unroll
      for (int i = 0; i < 2; ++i)
        a[i] = load_a(qbase + (size_t)(i * 16) * 1024 + kk, 1024, lane);
#pragma unroll
      for (int j = 0; j < 2; ++j)
        bfr[j] = load_b(kbase + (size_t)(t0 + j * 16) * 1024 + kk, 1024, lane);
#pragma unroll
      for (int i = 0; i < 2; ++i)
#pragma unroll
        for (int j = 0; j < 2; ++j) sacc[i][j] = wmma_bf16(a[i], bfr[j], sacc[i][j]);
    }

#pragma unroll
    for (int i = 0; i < 2; ++i)
#pragma unroll
      for (int j = 0; j < 2; ++j)
#pragma unroll
        for (int v = 0; v < 8; ++v)
          sS[c][msub * 32 + i * 16 + v + 8 * hl][j * 16 + nl] = sacc[i][j][v] * scale;
    __syncthreads();

    // ---- elementwise softmax across the 4 components + entropy ----
#pragma unroll
    for (int r = 0; r < 8; ++r) {
      int e = tid + r * 256;        // 0..2047 over 64x32 tile
      int m = e >> 5, n = e & 31;
      float x0 = sS[0][m][n], x1 = sS[1][m][n], x2 = sS[2][m][n], x3 = sS[3][m][n];
      float mx = fmaxf(fmaxf(x0, x1), fmaxf(x2, x3));
      float e0 = __expf(x0 - mx), e1 = __expf(x1 - mx);
      float e2 = __expf(x2 - mx), e3 = __expf(x3 - mx);
      float sum = e0 + e1 + e2 + e3;
      float inv = 1.0f / sum;
      float ls  = __logf(sum);
      float w0 = e0 * inv, w1 = e1 * inv, w2 = e2 * inv, w3 = e3 * inv;
      float ent = w0 * ((x0 - mx) - ls) + w1 * ((x1 - mx) - ls) +
                  w2 * ((x2 - mx) - ls) + w3 * ((x3 - mx) - ls);
      // entropy output is write-once, never re-read on device -> NT store
      __builtin_nontemporal_store(
          ent, ENT + ((size_t)(b * 16 + h) * 4096 + s0 + m) * 256 + t0 + n);
      sW[0][m][n] = f2bf(w0 * wscale);
      sW[1][m][n] = f2bf(w1 * wscale);
      sW[2][m][n] = f2bf(w2 * wscale);
      sW[3][m][n] = f2bf(w3 * wscale);
    }
    __syncthreads();

    // ---- out_c += w_c(32x32) @ V_c(32x64) ----
    {
      v16bf a2[2];
#pragma unroll
      for (int i = 0; i < 2; ++i)
        a2[i] = load_a(&sW[c][msub * 32 + i * 16][0], 32, lane);
#pragma unroll
      for (int j = 0; j < 4; ++j) {
        v16bf b2 = load_b(vbase + (size_t)(j * 16) * 256 + t0, 256, lane);
#pragma unroll
        for (int i = 0; i < 2; ++i) oacc[i][j] = wmma_bf16(a2[i], b2, oacc[i][j]);
      }
    }
    // next iteration's sS writes only occur after all threads pass the above
    // __syncthreads() (sS reads done); sW is only rewritten after the next barrier.
  }

  // store attention output (pre-Wo), bf16 row-major [bc][s][h*64+dh]
#pragma unroll
  for (int i = 0; i < 2; ++i)
#pragma unroll
    for (int j = 0; j < 4; ++j)
#pragma unroll
      for (int v = 0; v < 8; ++v) {
        int m = s0 + msub * 32 + i * 16 + v + 8 * hl;
        int n = h * 64 + j * 16 + nl;
        O[((size_t)bc * 4096 + m) * 1024 + n] = f2bf(oacc[i][j][v]);
      }
}

// ---------------------------------------------------------------------------
// Host launcher
// ---------------------------------------------------------------------------
extern "C" void kernel_launch(void* const* d_in, const int* in_sizes, int n_in,
                              void* d_out, int out_size, void* d_ws, size_t ws_size,
                              hipStream_t stream) {
  (void)in_sizes; (void)n_in; (void)out_size; (void)ws_size;
  const float* hs  = (const float*)d_in[0];  // [8,4096,1024]
  const float* ehs = (const float*)d_in[1];  // [8,256,2048]
  const float* Wq  = (const float*)d_in[2];  // [1024,1024]
  const float* Wk  = (const float*)d_in[3];  // [2048,1024]
  const float* Wv  = (const float*)d_in[4];  // [2048,1024]
  const float* Wo  = (const float*)d_in[5];  // [1024,1024]
  const float* bo  = (const float*)d_in[6];  // [1024]

  float* out = (float*)d_out;                      // [8,4096,1024]
  float* ent = out + (size_t)8 * 4096 * 1024;      // [2,16,4096,256]

  char* w = (char*)d_ws;
  auto carve = [&](size_t bytes) {
    char* p = w;
    w += (bytes + 255) & ~(size_t)255;
    return p;
  };
  unsigned short* hs_bf  = (unsigned short*)carve((size_t)33554432 * 2);
  unsigned short* ehs_bf = (unsigned short*)carve((size_t)4194304 * 2);
  unsigned short* WqT    = (unsigned short*)carve((size_t)1048576 * 2);
  unsigned short* WkT    = (unsigned short*)carve((size_t)2097152 * 2);
  unsigned short* WvT    = (unsigned short*)carve((size_t)2097152 * 2);
  unsigned short* WoT    = (unsigned short*)carve((size_t)1048576 * 2);
  unsigned short* q_bf   = (unsigned short*)carve((size_t)33554432 * 2);
  unsigned short* k_bf   = (unsigned short*)carve((size_t)2097152 * 2);
  unsigned short* v_bf   = (unsigned short*)carve((size_t)2097152 * 2);
  unsigned short* vT     = (unsigned short*)carve((size_t)2097152 * 2);
  unsigned short* attn   = (unsigned short*)carve((size_t)33554432 * 2);

  // 1) conversions (weights transposed into B^T layout)
  cvt_bf16_x8<<<4096, 256, 0, stream>>>(hs, hs_bf, 33554432 / 8);
  cvt_bf16_x8<<<2048, 256, 0, stream>>>(ehs, ehs_bf, 4194304 / 8);
  cvt_bf16_T<<<1024, 256, 0, stream>>>(Wq, WqT, 1024, 1024);
  cvt_bf16_T<<<2048, 256, 0, stream>>>(Wk, WkT, 2048, 1024);
  cvt_bf16_T<<<2048, 256, 0, stream>>>(Wv, WvT, 2048, 1024);
  cvt_bf16_T<<<1024, 256, 0, stream>>>(Wo, WoT, 1024, 1024);

  // 2) projections (block tile 128x256)
  gemm_bf16_out<<<dim3(4, 256), 256, 0, stream>>>(hs_bf, WqT, q_bf,
                                                  32768, 1024, 1024);
  gemm_bf16_out<<<dim3(4, 16), 256, 0, stream>>>(ehs_bf, WkT, k_bf,
                                                 2048, 1024, 2048);
  gemm_bf16_out<<<dim3(4, 16), 256, 0, stream>>>(ehs_bf, WvT, v_bf,
                                                 2048, 1024, 2048);
  transpose_v<<<8192, 256, 0, stream>>>(v_bf, vT);

  // 3) fused scores + component-softmax + entropy + attend
  fused_attn<<<dim3(64, 16, 2), 256, 0, stream>>>(q_bf, k_bf, vT, attn, ent);

  // 4) final projection with bias + residual epilogue (f32 output)
  gemm_f32_epi<<<dim3(4, 256), 256, 0, stream>>>(attn, WoT, out, bo, hs,
                                                 32768, 1024, 1024);
}
